// InjectorForSMPLX_74766790689504
// MI455X (gfx1250) — compile-verified, matrix operand order checked by reading, not run on hardware
//
#include <hip/hip_runtime.h>
#include <hip/hip_bf16.h>
#include <math.h>

// ---------------- problem constants (match reference) ----------------
#define BB   16
#define TASK 80
#define LQ   1024
#define DD   768
#define HH   12
#define HD   64
#define PP   4
#define NL   1
#define HF   64
#define WF   64
#define LIN  (HF * WF)      // 4096
#define EPSV 1e-6f

typedef __attribute__((ext_vector_type(16))) _Float16 v16h;
typedef __attribute__((ext_vector_type(8)))  float    v8f;
typedef __attribute__((ext_vector_type(4)))  unsigned v4u;
typedef __attribute__((ext_vector_type(8)))  int      v8i;
typedef __attribute__((ext_vector_type(4)))  int      v4i;
typedef __attribute__((ext_vector_type(4)))  unsigned v4ui;

#if __has_builtin(__builtin_amdgcn_tensor_load_to_lds) && __has_builtin(__builtin_amdgcn_s_wait_tensorcnt)
#define USE_TDM 1
#else
#define USE_TDM 0
#endif

// ---------------- f32 -> f16 convert ----------------
__global__ __launch_bounds__(256) void cvt_f16_kernel(const float* __restrict__ src,
                                                      _Float16* __restrict__ dst, int n) {
    int i = blockIdx.x * blockDim.x + threadIdx.x;
    if (i < n) dst[i] = (_Float16)src[i];
}

// ---------------- LayerNorm rows -> f16 ----------------
__global__ __launch_bounds__(256) void ln_f16_kernel(const float* __restrict__ x,
                                                     const float* __restrict__ w,
                                                     const float* __restrict__ bb,
                                                     _Float16* __restrict__ y,
                                                     int cols, int rowsPerBatch,
                                                     int batchStride, int rowOffset) {
    int r = blockIdx.x;
    int srcRow = (r / rowsPerBatch) * batchStride + rowOffset + (r % rowsPerBatch);
    const float* xr = x + (size_t)srcRow * cols;
    _Float16* yr = y + (size_t)r * cols;

    __shared__ float red[256];
    int tid = threadIdx.x;

    float s = 0.f;
    for (int c = tid; c < cols; c += 256) s += xr[c];
    red[tid] = s; __syncthreads();
    for (int o = 128; o > 0; o >>= 1) { if (tid < o) red[tid] += red[tid + o]; __syncthreads(); }
    float mu = red[0] / (float)cols;
    __syncthreads();

    float v = 0.f;
    for (int c = tid; c < cols; c += 256) { float d = xr[c] - mu; v += d * d; }
    red[tid] = v; __syncthreads();
    for (int o = 128; o > 0; o >>= 1) { if (tid < o) red[tid] += red[tid + o]; __syncthreads(); }
    float rstd = rsqrtf(red[0] / (float)cols + EPSV);

    for (int c = tid; c < cols; c += 256)
        yr[c] = (_Float16)((xr[c] - mu) * rstd * w[c] + bb[c]);
}

// ---------------- WMMA GEMM: C[M,N] = A[M,K]*B[K,N] + bias (K == 768) ----------
// Block tile 128x64, K-step 64, 256 threads = 8 waves (4x2), wave = 32x32.
// A tiles: TDM double-buffered in LDS [M][K_tile+8].
// B panel: staged ONCE per block, transposed, full-K resident: [N=64][768+8].
// Both fragment types are two contiguous 16B runs per lane -> ds_load_b128 pairs.
#define GT_M    128
#define GT_N    64
#define GT_K    64
#define GT_KMAX 768          // K of both GEMMs (compile-time for LDS panel)
#define APITCH  72           // 64 f16 + 8 pad (TDM codes: 32-DWORD rows, 4-DWORD pad)
#define BTPF    (GT_KMAX + 8)  // 776 f16 row pitch; 388 dwords == 4 mod 64 -> conflict-free

#if USE_TDM
// Build + issue a 2D TDM tensor_load_to_lds per CDNA5 ISA.
__device__ __forceinline__ void tdm_load_2d(const void* gaddr, unsigned ldsAddr,
                                            unsigned tensorW, unsigned tensorH,
                                            unsigned tileW, unsigned tileH,
                                            unsigned strideElems) {
    unsigned long long ga = (unsigned long long)(uintptr_t)gaddr;
    v4u g0;
    g0.x = 1u;                                   // count=1, user descriptor
    g0.y = ldsAddr;                              // lds_addr [63:32]
    g0.z = (unsigned)(ga & 0xffffffffu);         // global_addr lo
    g0.w = (unsigned)((ga >> 32) & 0x01ffffffu)  // global_addr [56:32]
         | 0x80000000u;                          // type=2 ("image") in [127:126]
    v8i g1;
    // w0: data_size=1(2B)<<16 | pad_enable<<20 | pad_interval(4:=32DW)<<22 | pad_amount(3:=4DW)<<25
    g1[0] = (int)((1u << 16) | (1u << 20) | (4u << 22) | (3u << 25));
    g1[1] = (int)((tensorW & 0xffffu) << 16);                                  // dim0 lo16 @63:48
    g1[2] = (int)(((tensorW >> 16) & 0xffffu) | ((tensorH & 0xffffu) << 16));  // dim0 hi / dim1 lo
    g1[3] = (int)(((tensorH >> 16) & 0xffffu) | ((tileW & 0xffffu) << 16));    // dim1 hi / tile_dim0
    g1[4] = (int)(tileH & 0xffffu);                                            // tile_dim1 (dim2=0)
    g1[5] = (int)strideElems;                                                  // dim0_stride lo32
    g1[6] = 0;
    g1[7] = 0;
    v4i z4 = {0, 0, 0, 0};
#if defined(__clang_major__) && __clang_major__ >= 23
    v8i z8 = {0, 0, 0, 0, 0, 0, 0, 0};
    __builtin_amdgcn_tensor_load_to_lds(g0, g1, z4, z4, z8, 0);
#else
    __builtin_amdgcn_tensor_load_to_lds(g0, g1, z4, z4, 0);
#endif
}

__device__ __forceinline__ unsigned lds_off(const void* p) {
    return (unsigned)(uintptr_t)p;   // LDS aperture: low 32 bits = LDS byte offset
}
#endif

// Two aligned 16B LDS reads -> one v16h fragment (no half-repacking).
__device__ __forceinline__ v16h load_frag(const _Float16* b0, const _Float16* b1) {
    struct Pair { v4ui lo, hi; } pr;
    pr.lo = *(const v4ui*)b0;
    pr.hi = *(const v4ui*)b1;
    return __builtin_bit_cast(v16h, pr);
}

// One K-stage (GT_K=64 -> 2 sub-steps of 32) of the 32x32 wave tile.
// Bt points at the k0 column of the resident transposed panel (row pitch BTPF).
__device__ __forceinline__ void tile_mma(const _Float16* As, const _Float16* Bt,
                                         int wm, int wn, int l16, int kb,
                                         v8f acc[2][2]) {
#pragma unroll
    for (int ks = 0; ks < GT_K; ks += 32) {
        v16h afrag[2], bfrag[2];
#pragma unroll
        for (int i = 0; i < 2; ++i) {
            const _Float16* ar = As + (size_t)(wm * 32 + i * 16 + l16) * APITCH + ks;
            afrag[i] = load_frag(ar + kb, ar + 16 + kb);
        }
#pragma unroll
        for (int j = 0; j < 2; ++j) {
            const _Float16* br = Bt + (size_t)(wn * 32 + j * 16 + l16) * BTPF + ks;
            bfrag[j] = load_frag(br + kb, br + 16 + kb);
        }
#pragma unroll
        for (int i = 0; i < 2; ++i)
#pragma unroll
            for (int j = 0; j < 2; ++j)
                acc[i][j] = __builtin_amdgcn_wmma_f32_16x16x32_f16(
                    false, afrag[i], false, bfrag[j],
                    (short)0, acc[i][j], false, false);
    }
}

template <typename OutT>
__global__ __launch_bounds__(256)
void gemm_wmma_kernel(const _Float16* __restrict__ A, const _Float16* __restrict__ Bm,
                      const float* __restrict__ bias, OutT* __restrict__ C,
                      int M, int N) {
    const int K = GT_KMAX;
#if USE_TDM
    __shared__ _Float16 As[2][GT_M][APITCH];   // 36,864 B, TDM double-buffered
#else
    __shared__ _Float16 As[1][GT_M][APITCH];
#endif
    __shared__ _Float16 Bt[GT_N][BTPF];        // 99,328 B, full-K transposed B panel

    const int tid  = threadIdx.x;
    const int lane = tid & 31;
    const int wave = tid >> 5;
    const int l16  = lane & 15;
    const int kb   = (lane >> 4) * 8;   // ISA 16-bit half-lane K offset
    const int wm   = wave >> 1;
    const int wn   = wave & 1;

    const int rowBase = blockIdx.y * GT_M;
    const int colBase = blockIdx.x * GT_N;
    const int S = K / GT_K;

    v8f acc[2][2] = {};

    // ---- stage the ENTIRE transposed B panel once ----
    // thread t: B row k = (chunk*64 + t>>2), 16 contiguous f16 at col (t&3)*16.
    const int bRow = tid >> 2;            // 0..63
    const int bCol = (tid & 3) * 16;      // 0..48
#pragma unroll
    for (int k0 = 0; k0 < GT_KMAX; k0 += GT_K) {
        const _Float16* src = &Bm[(size_t)(k0 + bRow) * N + colBase + bCol];
        _Float16 tmp[16];
        *(v4ui*)(tmp)     = *(const v4ui*)(src);
        *(v4ui*)(tmp + 8) = *(const v4ui*)(src + 8);
#pragma unroll
        for (int j = 0; j < 16; ++j) Bt[bCol + j][k0 + bRow] = tmp[j];
    }

#if USE_TDM
    if (wave == 0)   // async DMA of first A tile (TENSORcnt-tracked)
        tdm_load_2d(A + (size_t)rowBase * K, lds_off(&As[0][0][0]),
                    (unsigned)K, (unsigned)M, GT_K, GT_M, (unsigned)K);
    for (int s = 0; s < S; ++s) {
        const int buf = s & 1;
        if (wave == 0) {
            if (s + 1 < S) {   // DMA next A tile into the other buffer
                tdm_load_2d(A + (size_t)rowBase * K + (size_t)(s + 1) * GT_K,
                            lds_off(&As[buf ^ 1][0][0]), (unsigned)K, (unsigned)M,
                            GT_K, GT_M, (unsigned)K);
                __builtin_amdgcn_s_wait_tensorcnt(1);   // current A tile landed
            } else {
                __builtin_amdgcn_s_wait_tensorcnt(0);
            }
        }
        __syncthreads();   // A tile + (first iter) B panel visible to all waves
        tile_mma(&As[buf][0][0], &Bt[0][0] + s * GT_K, wm, wn, l16, kb, acc);
        __syncthreads();   // safe to overwrite the other A buffer
    }
#else
    const int aRow = tid >> 1;            // 0..127
    const int aCol = (tid & 1) * 32;      // 0 or 32
    for (int s = 0; s < S; ++s) {
        const int k0 = s * GT_K;
        {   // stage A tile (row-major, aligned 16B vector copies)
            const _Float16* src = &A[(size_t)(rowBase + aRow) * K + k0 + aCol];
            if (k0 + GT_K < K)
                __builtin_prefetch(&A[(size_t)(rowBase + aRow) * K + k0 + GT_K + aCol], 0, 1);
            v4ui* dst = (v4ui*)&As[0][aRow][aCol];
#pragma unroll
            for (int j = 0; j < 4; ++j) dst[j] = ((const v4ui*)src)[j];
        }
        __syncthreads();
        tile_mma(&As[0][0][0], &Bt[0][0] + k0, wm, wn, l16, kb, acc);
        __syncthreads();
    }
#endif

    // C layout: VGPR r, lanes 0-15 -> M=r, lanes 16-31 -> M=r+8, N=lane&15
    const int half = lane >> 4;
#pragma unroll
    for (int i = 0; i < 2; ++i) {
#pragma unroll
        for (int j = 0; j < 2; ++j) {
            const int col = colBase + wn * 32 + j * 16 + l16;
            const float bv = bias[col];
#pragma unroll
            for (int r = 0; r < 8; ++r) {
                const int row = rowBase + wm * 32 + i * 16 + half * 8 + r;
                C[(size_t)row * N + col] = (OutT)(acc[i][j][r] + bv);
            }
        }
    }
}

// ---------------- off / attention-weight projections ----------------
__global__ __launch_bounds__(256)
void offaw_kernel(const _Float16* __restrict__ qn,
                  const float* __restrict__ W_off, const float* __restrict__ b_off,
                  const float* __restrict__ W_attw, const float* __restrict__ b_attw,
                  float* __restrict__ off, float* __restrict__ awl) {
    const int NOUT = HH * PP * 2 + HH * PP;   // 96 + 48 = 144
    int idx = blockIdx.x * blockDim.x + threadIdx.x;
    int total = BB * LQ * NOUT;
    if (idx >= total) return;
    int o  = idx % NOUT;
    int bq = idx / NOUT;
    const _Float16* qrow = qn + (size_t)bq * DD;

    if (o < HH * PP * 2) {
        float s = b_off[o];
        for (int k = 0; k < DD; ++k) s += (float)qrow[k] * W_off[(size_t)k * (HH * PP * 2) + o];
        off[(size_t)bq * (HH * PP * 2) + o] = s;
    } else {
        int o2 = o - HH * PP * 2;
        float s = b_attw[o2];
        for (int k = 0; k < DD; ++k) s += (float)qrow[k] * W_attw[(size_t)k * (HH * PP) + o2];
        awl[(size_t)bq * (HH * PP) + o2] = s;
    }
}

// ---------------- bilinear sampling + softmax-weighted sum ----------------
__global__ __launch_bounds__(256)
void sample_kernel(const _Float16* __restrict__ value,   // [B*LIN, D]
                   const float* __restrict__ refp,       // [B,LQ,NL,2]
                   const float* __restrict__ off,        // [B*LQ, HH*P*2]
                   const float* __restrict__ awl,        // [B*LQ, HH*P]
                   _Float16* __restrict__ samp) {        // [B*LQ, D]
    int gwave = (blockIdx.x * blockDim.x + threadIdx.x) >> 5;
    int lane  = threadIdx.x & 31;
    int total = BB * LQ * HH;
    if (gwave >= total) return;
    int h  = gwave % HH;
    int bq = gwave / HH;
    int b  = bq / LQ;

    float lg[PP];
    float mx = -1e30f;
#pragma unroll
    for (int p = 0; p < PP; ++p) {
        lg[p] = awl[(size_t)bq * (HH * PP) + h * PP + p];
        mx = fmaxf(mx, lg[p]);
    }
    float sum = 0.f;
#pragma unroll
    for (int p = 0; p < PP; ++p) { lg[p] = __expf(lg[p] - mx); sum += lg[p]; }
    float inv = 1.0f / sum;

    float rx = refp[((size_t)bq * NL) * 2 + 0];
    float ry = refp[((size_t)bq * NL) * 2 + 1];

    int d0 = lane * 2;
    float acc0 = 0.f, acc1 = 0.f;
#pragma unroll
    for (int p = 0; p < PP; ++p) {
        float wp = lg[p] * inv;
        float ox = off[(size_t)bq * (HH * PP * 2) + (h * PP + p) * 2 + 0];
        float oy = off[(size_t)bq * (HH * PP * 2) + (h * PP + p) * 2 + 1];
        float x = (rx + ox / (float)WF) * (float)WF - 0.5f;
        float y = (ry + oy / (float)HF) * (float)HF - 0.5f;
        float x0 = floorf(x), y0 = floorf(y);
#pragma unroll
        for (int dx = 0; dx < 2; ++dx) {
#pragma unroll
            for (int dy = 0; dy < 2; ++dy) {
                float xc = x0 + (float)dx, yc = y0 + (float)dy;
                float w = (1.0f - fabsf(x - xc)) * (1.0f - fabsf(y - yc));
                bool valid = (xc >= 0.f) && (xc <= (float)(WF - 1)) &&
                             (yc >= 0.f) && (yc <= (float)(HF - 1));
                w = valid ? w : 0.0f;
                if (w != 0.0f) {
                    int xi = (int)fminf(fmaxf(xc, 0.f), (float)(WF - 1));
                    int yi = (int)fminf(fmaxf(yc, 0.f), (float)(HF - 1));
                    int linp = yi * WF + xi;
                    const _Float16* vp = value + ((size_t)(b * LIN + linp) * DD + h * HD + d0);
                    float ww = wp * w;
                    acc0 += ww * (float)vp[0];
                    acc1 += ww * (float)vp[1];
                }
            }
        }
    }
    _Float16* dst = samp + (size_t)bq * DD + h * HD + d0;
    dst[0] = (_Float16)acc0;
    dst[1] = (_Float16)acc1;
}

// ---------------- residual + concat ----------------
__global__ __launch_bounds__(256)
void final_kernel(const float* __restrict__ query,
                  const float* __restrict__ attn,
                  const float* __restrict__ gamma,
                  float* __restrict__ out) {
    const int ROWS = TASK + LQ;
    size_t idx = (size_t)blockIdx.x * blockDim.x + threadIdx.x;
    size_t total = (size_t)BB * ROWS * DD;
    if (idx >= total) return;
    int d = (int)(idx % DD);
    size_t row = idx / DD;
    int l = (int)(row % ROWS);
    int b = (int)(row / ROWS);
    float qv = query[idx];
    if (l < TASK) {
        out[idx] = qv;
    } else {
        int qi = l - TASK;
        out[idx] = qv + gamma[d] * attn[((size_t)b * LQ + qi) * DD + d];
    }
}

// ---------------- host launcher ----------------
extern "C" void kernel_launch(void* const* d_in, const int* in_sizes, int n_in,
                              void* d_out, int out_size, void* d_ws, size_t ws_size,
                              hipStream_t stream) {
    const float* query  = (const float*)d_in[0];
    const float* refp   = (const float*)d_in[1];
    const float* feat   = (const float*)d_in[2];
    const float* qn_w   = (const float*)d_in[5];
    const float* qn_b   = (const float*)d_in[6];
    const float* fn_w   = (const float*)d_in[7];
    const float* fn_b   = (const float*)d_in[8];
    const float* gamma  = (const float*)d_in[9];
    const float* W_off  = (const float*)d_in[10];
    const float* b_off  = (const float*)d_in[11];
    const float* W_attw = (const float*)d_in[12];
    const float* b_attw = (const float*)d_in[13];
    const float* W_val  = (const float*)d_in[14];
    const float* b_val  = (const float*)d_in[15];
    const float* W_out  = (const float*)d_in[16];
    const float* b_out  = (const float*)d_in[17];

    char* p = (char*)d_ws;
    auto alloc = [&](size_t bytes) {
        void* r = (void*)p;
        p += (bytes + 255) & ~(size_t)255;
        return r;
    };
    _Float16* fn_h    = (_Float16*)alloc((size_t)BB * LIN * DD * 2);
    _Float16* qn_h    = (_Float16*)alloc((size_t)BB * LQ * DD * 2);
    _Float16* Wval_h  = (_Float16*)alloc((size_t)DD * DD * 2);
    _Float16* Wout_h  = (_Float16*)alloc((size_t)DD * DD * 2);
    _Float16* value_h = (_Float16*)alloc((size_t)BB * LIN * DD * 2);
    float*    off_f   = (float*)alloc((size_t)BB * LQ * HH * PP * 2 * 4);
    float*    aw_f    = (float*)alloc((size_t)BB * LQ * HH * PP * 4);
    _Float16* samp_h  = (_Float16*)alloc((size_t)BB * LQ * DD * 2);
    float*    attn_f  = (float*)alloc((size_t)BB * LQ * DD * 4);
    (void)ws_size;

    {
        int n = DD * DD;
        cvt_f16_kernel<<<(n + 255) / 256, 256, 0, stream>>>(W_val, Wval_h, n);
        cvt_f16_kernel<<<(n + 255) / 256, 256, 0, stream>>>(W_out, Wout_h, n);
    }
    ln_f16_kernel<<<BB * LIN, 256, 0, stream>>>(feat, fn_w, fn_b, fn_h, DD, LIN, LIN, 0);
    ln_f16_kernel<<<BB * LQ, 256, 0, stream>>>(query, qn_w, qn_b, qn_h, DD, LQ, TASK + LQ, TASK);

    {   // value = fn @ W_val + b_val  -> f16   (65536 x 768 x 768)
        dim3 grid(DD / GT_N, (BB * LIN) / GT_M);
        gemm_wmma_kernel<_Float16><<<grid, 256, 0, stream>>>(fn_h, Wval_h, b_val, value_h,
                                                             BB * LIN, DD);
    }
    {
        int total = BB * LQ * (HH * PP * 2 + HH * PP);
        offaw_kernel<<<(total + 255) / 256, 256, 0, stream>>>(qn_h, W_off, b_off,
                                                              W_attw, b_attw, off_f, aw_f);
    }
    {
        int waves = BB * LQ * HH;
        sample_kernel<<<(waves * 32 + 255) / 256, 256, 0, stream>>>(value_h, refp, off_f,
                                                                    aw_f, samp_h);
    }
    {   // attn = samp @ W_out + b_out -> f32   (16384 x 768 x 768)
        dim3 grid(DD / GT_N, (BB * LQ) / GT_M);
        gemm_wmma_kernel<float><<<grid, 256, 0, stream>>>(samp_h, Wout_h, b_out, attn_f,
                                                          BB * LQ, DD);
    }
    {
        size_t total = (size_t)BB * (TASK + LQ) * DD;
        final_kernel<<<(unsigned)((total + 255) / 256), 256, 0, stream>>>(query, attn_f,
                                                                          gamma, (float*)d_out);
    }
}
